// HierarchicalReasoningNetwork_81939386073363
// MI455X (gfx1250) — matrix operand order repaired and checked by reading, not compiled, not run on hardware
//
#include <hip/hip_runtime.h>

#define NN   2048
#define OBS  128
#define HIDD 256

typedef __bf16 bf16_t;
typedef __attribute__((ext_vector_type(8)))  __bf16 v8bf;
typedef __attribute__((ext_vector_type(16))) __bf16 v16bf;
typedef __attribute__((ext_vector_type(8)))  float  v8f;
typedef __attribute__((ext_vector_type(4)))  int    v4i_t;

enum { FLAG_BIAS = 1, FLAG_RELU = 2, FLAG_ROWSCALE = 4, FLAG_SELECT = 8 };

// CDNA5 async global->LDS copy path (ASYNCcnt-tracked), with sync fallback.
#if defined(__HIP_DEVICE_COMPILE__) && \
    __has_builtin(__builtin_amdgcn_global_load_async_to_lds_b128) && \
    __has_builtin(__builtin_amdgcn_s_wait_asynccnt)
#define USE_ASYNC_LDS 1
#else
#define USE_ASYNC_LDS 0
#endif

#if USE_ASYNC_LDS
__device__ __forceinline__ void async_cp16(const bf16_t* g, bf16_t* l) {
  // prototype (from compiler diagnostic): (v4i* global_src, v4i* lds_dst, int offset, int cpol)
  __builtin_amdgcn_global_load_async_to_lds_b128((v4i_t*)g, (v4i_t*)l, 0, 0);
}
#endif

#define LDSPAD 40   // 32 data + 8 pad bf16 -> 80B row stride, 16B-aligned cols

// ---------------------------------------------------------------------------
// Generic bf16 WMMA GEMM: out(M x 256) = epilogue(A(M x K) @ Bt(256 x K)^T)
// A row-major bf16 (lda), Bt is B transposed, row-major bf16 (ldb).
// 128 threads = 4 waves; block tile 64(M) x 64(N); each wave computes 32x32.
// K-loop double-buffers LDS via async copies when available.
// ---------------------------------------------------------------------------
__global__ __launch_bounds__(128) void wmma_gemm(
    const bf16_t* __restrict__ A, int lda,
    const bf16_t* __restrict__ Bt, int ldb, int K,
    const float* __restrict__ bias,
    const float* __restrict__ rowscale,   // invdeg (FLAG_ROWSCALE)
    const float* __restrict__ deg,        // deg    (FLAG_SELECT)
    const float* __restrict__ fallback, int ldf,
    float* __restrict__ outF, int ldoF,
    bf16_t* __restrict__ outB, int ldoB,
    int flags)
{
  __shared__ __align__(16) bf16_t As[2][64][LDSPAD];
  __shared__ __align__(16) bf16_t Bs[2][64][LDSPAD];

  const int tid    = threadIdx.x;
  const int wave   = tid >> 5;
  const int lane   = tid & 31;
  const int hf     = lane >> 4;     // lane half (0/1)
  const int l16    = lane & 15;
  const int waveM  = (wave >> 1) * 32;
  const int waveN  = (wave & 1) * 32;
  const int blockM = blockIdx.y * 64;
  const int blockN = blockIdx.x * 64;

  // staging segment for this thread: 4 16B segments (2 A + 2 B) per tile
  const int sr0 = (tid * 2) >> 2, sc0 = ((tid * 2) & 3) * 8;
  const int sr1 = (tid * 2 + 1) >> 2, sc1 = ((tid * 2 + 1) & 3) * 8;

  v8f acc[2][2];
#pragma unroll
  for (int mi = 0; mi < 2; ++mi)
#pragma unroll
    for (int ni = 0; ni < 2; ++ni)
      acc[mi][ni] = 0.0f;

  const int nk = K >> 5;

#if USE_ASYNC_LDS
  auto stage = [&](int buf, int k0) {
    async_cp16(&A [(size_t)(blockM + sr0) * lda + k0 + sc0], &As[buf][sr0][sc0]);
    async_cp16(&Bt[(size_t)(blockN + sr0) * ldb + k0 + sc0], &Bs[buf][sr0][sc0]);
    async_cp16(&A [(size_t)(blockM + sr1) * lda + k0 + sc1], &As[buf][sr1][sc1]);
    async_cp16(&Bt[(size_t)(blockN + sr1) * ldb + k0 + sc1], &Bs[buf][sr1][sc1]);
  };
  stage(0, 0);
#else
  auto stage = [&](int buf, int k0) {
    *(v8bf*)&As[buf][sr0][sc0] = *(const v8bf*)&A [(size_t)(blockM + sr0) * lda + k0 + sc0];
    *(v8bf*)&Bs[buf][sr0][sc0] = *(const v8bf*)&Bt[(size_t)(blockN + sr0) * ldb + k0 + sc0];
    *(v8bf*)&As[buf][sr1][sc1] = *(const v8bf*)&A [(size_t)(blockM + sr1) * lda + k0 + sc1];
    *(v8bf*)&Bs[buf][sr1][sc1] = *(const v8bf*)&Bt[(size_t)(blockN + sr1) * ldb + k0 + sc1];
  };
#endif

  for (int t = 0; t < nk; ++t) {
#if USE_ASYNC_LDS
    const int buf = t & 1;
    if (t + 1 < nk) {                       // prefetch next tile, retire current
      stage(buf ^ 1, (t + 1) << 5);
      __builtin_amdgcn_s_wait_asynccnt(4);  // 4 older ops (tile t) complete, in order
    } else {
      __builtin_amdgcn_s_wait_asynccnt(0);
    }
#else
    const int buf = 0;
    stage(0, t << 5);
#endif
    __syncthreads();

    // A fragments: 16x32, lane row = M, K runs {8h..8h+7} and {16+8h..+7}
    v16bf af[2];
#pragma unroll
    for (int mi = 0; mi < 2; ++mi) {
      int r = waveM + mi * 16 + l16;
      union { struct { v8bf lo, hi; } p; v16bf v; } u;
      u.p.lo = *(const v8bf*)&As[buf][r][8 * hf];
      u.p.hi = *(const v8bf*)&As[buf][r][16 + 8 * hf];
      af[mi] = u.v;
    }
    // B fragments: 32x16, lane row = N in Bt, K runs {8h..8h+7} and {16+8h..+7}
    v16bf bfr[2];
#pragma unroll
    for (int ni = 0; ni < 2; ++ni) {
      int r = waveN + ni * 16 + l16;
      union { struct { v8bf lo, hi; } p; v16bf v; } u;
      u.p.lo = *(const v8bf*)&Bs[buf][r][8 * hf];
      u.p.hi = *(const v8bf*)&Bs[buf][r][16 + 8 * hf];
      bfr[ni] = u.v;
    }
#pragma unroll
    for (int mi = 0; mi < 2; ++mi)
#pragma unroll
      for (int ni = 0; ni < 2; ++ni)
        acc[mi][ni] = __builtin_amdgcn_wmma_f32_16x16x32_bf16(
            false, af[mi], false, bfr[ni], (short)0, acc[mi][ni], false, false);
    __syncthreads();
  }

  // Epilogue. C/D layout: VGPR r -> M = r + 8*half, N = lane&15.
  const int n0 = blockN + waveN + l16;
  float biasv[2] = {0.0f, 0.0f};
  if (flags & FLAG_BIAS) { biasv[0] = bias[n0]; biasv[1] = bias[n0 + 16]; }
#pragma unroll
  for (int mi = 0; mi < 2; ++mi) {
#pragma unroll
    for (int r = 0; r < 8; ++r) {
      const int m = blockM + waveM + mi * 16 + hf * 8 + r;
      float rs = 1.0f;
      if (flags & FLAG_ROWSCALE) rs = rowscale[m];
      bool takeFB = false;
      if (flags & FLAG_SELECT) takeFB = (deg[m] <= 0.0f);
#pragma unroll
      for (int ni = 0; ni < 2; ++ni) {
        const int n = n0 + ni * 16;
        float v = (acc[mi][ni][r] + biasv[ni]) * rs;
        if (flags & FLAG_RELU) v = fmaxf(v, 0.0f);
        if (takeFB) v = fallback[(size_t)m * ldf + n];
        if (outF) outF[(size_t)m * ldoF + n] = v;
        if (outB) outB[(size_t)m * ldoB + n] = (bf16_t)v;
      }
    }
  }
}

// ---------------------------------------------------------------------------
// Helpers
// ---------------------------------------------------------------------------
__global__ __launch_bounds__(256) void cast_f32_bf16(const float* __restrict__ in,
                                                     bf16_t* __restrict__ out, int n) {
  int i = blockIdx.x * 256 + threadIdx.x;
  if (i < n) out[i] = (bf16_t)in[i];
}

// out(C x R) bf16 = transpose(in(R x C) f32). R, C multiples of 32.
__global__ __launch_bounds__(256) void transpose_cast(const float* __restrict__ in,
                                                      bf16_t* __restrict__ out,
                                                      int R, int C) {
  __shared__ float t[32][33];
  int bx = blockIdx.x * 32, by = blockIdx.y * 32;
  int tx = threadIdx.x & 31, ty = threadIdx.x >> 5;
#pragma unroll
  for (int i = 0; i < 32; i += 8)
    t[ty + i][tx] = in[(size_t)(by + ty + i) * C + bx + tx];
  __syncthreads();
#pragma unroll
  for (int i = 0; i < 32; i += 8)
    out[(size_t)(bx + ty + i) * R + by + tx] = (bf16_t)t[tx][ty + i];
}

// mask (bf16 0/1, zero diagonal), deg, 1/max(deg,1) from adj. One block per row.
__global__ __launch_bounds__(256) void mask_deg(const int* __restrict__ adj,
                                                bf16_t* __restrict__ mask,
                                                float* __restrict__ degF,
                                                float* __restrict__ invd) {
  __shared__ float red[256];
  int row = blockIdx.x, tid = threadIdx.x;
  float cnt = 0.0f;
  for (int j = tid; j < NN; j += 256) {
    int a = adj[(size_t)row * NN + j];
    float mv = (a > 0 && j != row) ? 1.0f : 0.0f;
    mask[(size_t)row * NN + j] = (bf16_t)mv;
    cnt += mv;
  }
  red[tid] = cnt; __syncthreads();
  for (int s = 128; s > 0; s >>= 1) { if (tid < s) red[tid] += red[tid + s]; __syncthreads(); }
  if (tid == 0) { degF[row] = red[0]; invd[row] = 1.0f / fmaxf(red[0], 1.0f); }
}

// glob stage 1: partial dot of flattened indiv with gw1 rows. 2048 blocks x 256 rows.
// This is the roofline term: one coalesced pass over 536MB of gw1 (f32).
__global__ __launch_bounds__(256) void glob_partial(const float* __restrict__ v,
                                                    const float* __restrict__ gw1,
                                                    float* __restrict__ part) {
  __shared__ float vch[256];
  int t = threadIdx.x;
  int r0 = blockIdx.x * 256;
  vch[t] = v[r0 + t];
  __syncthreads();
  float acc = 0.0f;
#pragma unroll 8
  for (int i = 0; i < 256; ++i)
    acc += vch[i] * gw1[(size_t)(r0 + i) * HIDD + t];
  part[(size_t)blockIdx.x * HIDD + t] = acc;
}

// two-stage deterministic reduce of the 2048 partial rows
__global__ __launch_bounds__(256) void glob_reduce1(const float* __restrict__ part,
                                                    float* __restrict__ r1) {
  int t = threadIdx.x, b = blockIdx.x;         // 64 blocks x 32 partials
  float s = 0.0f;
#pragma unroll 8
  for (int j = 0; j < 32; ++j)
    s += part[(size_t)(b * 32 + j) * HIDD + t];
  r1[(size_t)b * HIDD + t] = s;
}

__global__ __launch_bounds__(256) void glob_reduce2(const float* __restrict__ r1,
                                                    const float* __restrict__ gb1,
                                                    float* __restrict__ h1) {
  int t = threadIdx.x;
  float s = gb1[t];
#pragma unroll 8
  for (int b = 0; b < 64; ++b) s += r1[(size_t)b * HIDD + t];
  h1[t] = fmaxf(s, 0.0f);
}

__global__ __launch_bounds__(256) void glob_mv2(const float* __restrict__ h1,
                                                const float* __restrict__ gw2,
                                                const float* __restrict__ gb2,
                                                float* __restrict__ g) {
  __shared__ float hs[HIDD];
  int t = threadIdx.x;
  hs[t] = h1[t];
  __syncthreads();
  float s = gb2[t];
  for (int k = 0; k < HIDD; ++k) s += hs[k] * gw2[(size_t)k * HIDD + t];
  g[t] = s;
}

// fused[m][512 + n] = bf16(glob[n]); one block per row.
__global__ __launch_bounds__(256) void glob_bcast(const float* __restrict__ g,
                                                  bf16_t* __restrict__ fused) {
  int t = threadIdx.x;
  fused[(size_t)blockIdx.x * 768 + 512 + t] = (bf16_t)g[t];
}

// fused[m][0..255] = bf16(indivF[m][0..255]); one block per row.
__global__ __launch_bounds__(256) void indiv_to_fused(const float* __restrict__ src,
                                                      bf16_t* __restrict__ fused) {
  int t = threadIdx.x;
  fused[(size_t)blockIdx.x * 768 + t] = (bf16_t)src[(size_t)blockIdx.x * HIDD + t];
}

// ---------------------------------------------------------------------------
extern "C" void kernel_launch(void* const* d_in, const int* in_sizes, int n_in,
                              void* d_out, int out_size, void* d_ws, size_t ws_size,
                              hipStream_t stream) {
  const float* obs = (const float*)d_in[0];
  const int*   adj = (const int*)d_in[1];
  const float* iw1 = (const float*)d_in[2];
  const float* ib1 = (const float*)d_in[3];
  const float* iw2 = (const float*)d_in[4];
  const float* ib2 = (const float*)d_in[5];
  const float* lw1 = (const float*)d_in[6];
  const float* lb1 = (const float*)d_in[7];
  const float* lw2 = (const float*)d_in[8];
  const float* lb2 = (const float*)d_in[9];
  const float* gw1 = (const float*)d_in[10];
  const float* gb1 = (const float*)d_in[11];
  const float* gw2 = (const float*)d_in[12];
  const float* gb2 = (const float*)d_in[13];
  const float* fw1 = (const float*)d_in[14];
  const float* fb1 = (const float*)d_in[15];
  const float* fw2 = (const float*)d_in[16];
  const float* fb2 = (const float*)d_in[17];
  float* out = (float*)d_out;

  char* ws = (char*)d_ws;
  size_t off = 0;
  auto alloc = [&](size_t bytes) -> char* {
    char* p = ws + off;
    off += (bytes + 255) & ~(size_t)255;
    return p;
  };

  bf16_t* obsB   = (bf16_t*)alloc((size_t)NN * OBS * 2);
  bf16_t* iw1t   = (bf16_t*)alloc((size_t)HIDD * OBS * 2);
  bf16_t* iw2t   = (bf16_t*)alloc((size_t)HIDD * HIDD * 2);
  bf16_t* lw1t   = (bf16_t*)alloc((size_t)HIDD * 512 * 2);
  bf16_t* lw2t   = (bf16_t*)alloc((size_t)HIDD * HIDD * 2);
  bf16_t* fw1t   = (bf16_t*)alloc((size_t)HIDD * 768 * 2);
  bf16_t* fw2t   = (bf16_t*)alloc((size_t)HIDD * HIDD * 2);
  bf16_t* H1i    = (bf16_t*)alloc((size_t)NN * HIDD * 2);
  float*  indivF = (float*) alloc((size_t)NN * HIDD * 4);
  bf16_t* indivT = (bf16_t*)alloc((size_t)HIDD * NN * 2);
  bf16_t* maskB  = (bf16_t*)alloc((size_t)NN * NN * 2);
  float*  degF   = (float*) alloc((size_t)NN * 4);
  float*  invd   = (float*) alloc((size_t)NN * 4);
  bf16_t* catL   = (bf16_t*)alloc((size_t)NN * 512 * 2);   // [indiv | neigh_mean]
  bf16_t* L1     = (bf16_t*)alloc((size_t)NN * HIDD * 2);
  bf16_t* fused  = (bf16_t*)alloc((size_t)NN * 768 * 2);   // [indiv | local | glob]
  bf16_t* F1     = (bf16_t*)alloc((size_t)NN * HIDD * 2);
  float*  gpart  = (float*) alloc((size_t)NN * HIDD * 4);
  float*  gr1    = (float*) alloc((size_t)64 * HIDD * 4);
  float*  h1     = (float*) alloc((size_t)HIDD * 4);
  float*  globv  = (float*) alloc((size_t)HIDD * 4);
  (void)ws_size; (void)in_sizes; (void)n_in; (void)out_size;

  dim3 blk(128);
  dim3 grd(256 / 64, NN / 64);   // all GEMMs: M=2048, N=256

  // --- prep: casts / transposes / mask+deg -------------------------------
  cast_f32_bf16<<<(NN * OBS + 255) / 256, 256, 0, stream>>>(obs, obsB, NN * OBS);
  transpose_cast<<<dim3(HIDD / 32, OBS / 32),  256, 0, stream>>>(iw1, iw1t, OBS,  HIDD);
  transpose_cast<<<dim3(HIDD / 32, HIDD / 32), 256, 0, stream>>>(iw2, iw2t, HIDD, HIDD);
  transpose_cast<<<dim3(HIDD / 32, 512 / 32),  256, 0, stream>>>(lw1, lw1t, 512,  HIDD);
  transpose_cast<<<dim3(HIDD / 32, HIDD / 32), 256, 0, stream>>>(lw2, lw2t, HIDD, HIDD);
  transpose_cast<<<dim3(HIDD / 32, 768 / 32),  256, 0, stream>>>(fw1, fw1t, 768,  HIDD);
  transpose_cast<<<dim3(HIDD / 32, HIDD / 32), 256, 0, stream>>>(fw2, fw2t, HIDD, HIDD);
  mask_deg<<<NN, 256, 0, stream>>>(adj, maskB, degF, invd);

  // --- indiv MLP ---------------------------------------------------------
  wmma_gemm<<<grd, blk, 0, stream>>>(obsB, OBS, iw1t, OBS, OBS, ib1,
      nullptr, nullptr, nullptr, 0, nullptr, 0, H1i, HIDD, FLAG_BIAS | FLAG_RELU);
  wmma_gemm<<<grd, blk, 0, stream>>>(H1i, HIDD, iw2t, HIDD, HIDD, ib2,
      nullptr, nullptr, nullptr, 0, indivF, HIDD, catL, 512, FLAG_BIAS);

  // --- glob branch (memory-bound: one pass over 536MB gw1) ---------------
  glob_partial<<<NN, 256, 0, stream>>>(indivF, gw1, gpart);
  glob_reduce1<<<64, 256, 0, stream>>>(gpart, gr1);
  glob_reduce2<<<1, 256, 0, stream>>>(gr1, gb1, h1);
  glob_mv2<<<1, 256, 0, stream>>>(h1, gw2, gb2, globv);

  // --- neighbor mean: mask @ indiv (bf16 WMMA, K=2048) -------------------
  transpose_cast<<<dim3(HIDD / 32, NN / 32), 256, 0, stream>>>(indivF, indivT, NN, HIDD);
  wmma_gemm<<<grd, blk, 0, stream>>>(maskB, NN, indivT, NN, NN, nullptr,
      invd, nullptr, nullptr, 0, nullptr, 0, catL + 256, 512, FLAG_ROWSCALE);

  // --- local MLP (with deg==0 fallback fused into epilogue) --------------
  wmma_gemm<<<grd, blk, 0, stream>>>(catL, 512, lw1t, 512, 512, lb1,
      nullptr, nullptr, nullptr, 0, nullptr, 0, L1, HIDD, FLAG_BIAS | FLAG_RELU);
  wmma_gemm<<<grd, blk, 0, stream>>>(L1, HIDD, lw2t, HIDD, HIDD, lb2,
      nullptr, degF, indivF, HIDD, nullptr, 0, fused + 256, 768, FLAG_BIAS | FLAG_SELECT);

  // --- assemble fused = [indiv | local | glob] ---------------------------
  indiv_to_fused<<<NN, 256, 0, stream>>>(indivF, fused);
  glob_bcast<<<NN, 256, 0, stream>>>(globv, fused);

  // --- fused MLP -> output ----------------------------------------------
  wmma_gemm<<<grd, blk, 0, stream>>>(fused, 768, fw1t, 768, 768, fb1,
      nullptr, nullptr, nullptr, 0, nullptr, 0, F1, HIDD, FLAG_BIAS | FLAG_RELU);
  wmma_gemm<<<grd, blk, 0, stream>>>(F1, HIDD, fw2t, HIDD, HIDD, fb2,
      nullptr, nullptr, nullptr, 0, out, HIDD, nullptr, 0, FLAG_BIAS);
}
